// SingleHeadAttention_85066122265669
// MI455X (gfx1250) — compile-verified, matrix operand order checked by reading, not run on hardware
//
#include <hip/hip_runtime.h>

// ---------------------------------------------------------------------------
// Single-head attention for MI455X (gfx1250), compile-only tuned.
// bf16 WMMA GEMMs (v_wmma_f32_16x16x32_bf16), 128x128 block tiles, 8 waves,
// per-lane-contiguous LDS fragment layout (2x ds_load_b128 per fragment),
// double-buffered staging.  Data movement showcases all gfx1250 paths:
//   - scores kernel: TDM tensor_load_to_lds (TENSORcnt), one D# per tile,
//     using TDM LDS padding (4 dwords per 16) to hit the padded tile layout
//   - PV kernel A tile: global_load_async_to_lds_b128 (ASYNCcnt, inline asm)
//   - proj kernel: register staging (f32->bf16 conversion en route)
// ---------------------------------------------------------------------------

// ---- feature probes: results appear in compiler stderr ---------------------
#if __has_builtin(__builtin_amdgcn_tensor_load_to_lds)
#pragma message("PROBE: has __builtin_amdgcn_tensor_load_to_lds")
#endif
#if __has_builtin(__builtin_amdgcn_s_wait_tensorcnt)
#pragma message("PROBE: has __builtin_amdgcn_s_wait_tensorcnt")
#endif

typedef __attribute__((ext_vector_type(16))) __bf16        v16bf;
typedef __attribute__((ext_vector_type(8)))  float         v8f;
typedef __attribute__((ext_vector_type(8)))  unsigned int  v8u;
typedef __attribute__((ext_vector_type(4)))  unsigned int  v4u;
typedef __attribute__((ext_vector_type(8)))  int           v8i;
typedef __attribute__((ext_vector_type(4)))  int           v4i;

#define DIM    1024
#define BATCH  4
#define SEQ    2048
#define MROWS  (BATCH * SEQ)

#define MT   128         // M tile per workgroup
#define NT   128         // N tile per workgroup
#define KT   32          // K chunk (one bf16 WMMA depth)
#define TSTR 20          // LDS tile row stride in dwords (16 data + 4 pad)

#if __has_builtin(__builtin_amdgcn_tensor_load_to_lds) && \
    __has_builtin(__builtin_amdgcn_s_wait_tensorcnt)
#define HAS_TDM 1
#else
#define HAS_TDM 0
#endif

// ---- helpers ---------------------------------------------------------------

__device__ __forceinline__ unsigned int pack2_bf16(float x, float y) {
  unsigned int ux = __builtin_bit_cast(unsigned int, x);
  unsigned int uy = __builtin_bit_cast(unsigned int, y);
  ux += 0x7fffu + ((ux >> 16) & 1u);   // round-to-nearest-even
  uy += 0x7fffu + ((uy >> 16) & 1u);
  return (ux >> 16) | (uy & 0xffff0000u);
}
__device__ __forceinline__ unsigned short f2bf(float x) {
  unsigned int u = __builtin_bit_cast(unsigned int, x);
  u += 0x7fffu + ((u >> 16) & 1u);
  return (unsigned short)(u >> 16);
}

// gfx1250 async memory->LDS copy (ASYNCcnt): 16B per lane, per-lane LDS dest.
__device__ __forceinline__ void async_copy_b128(const void* gptr, void* lptr) {
  unsigned int       lds = (unsigned int)(unsigned long long)lptr;  // LDS offset
  unsigned long long ga  = (unsigned long long)gptr;
  asm volatile("global_load_async_to_lds_b128 %0, %1, off"
               :: "v"(lds), "v"(ga) : "memory");
}
__device__ __forceinline__ void wait_async0() {
  asm volatile("s_wait_asynccnt 0" ::: "memory");
}

#if HAS_TDM
// One TDM descriptor loads a [tile_d1 rows x tile_d0 bf16] tile (row stride
// stride0 elements) into LDS, inserting 4 pad dwords after every 16 data
// dwords -> exactly the TSTR=20 padded tile layout.  D# bitfields per CDNA5
// ISA 8.3/8.4 (group0: count/lds/global/type; group1: flags/dims/strides).
// This toolchain's builtin takes 6 args (g0 v4u, g1 v8i, g2 v4i, g3 v4i,
// extra v8i, cpol) -- trailing groups zero for a 2D tensor.
__device__ __forceinline__ void tdm_load_2d(void* lptr, const void* gptr,
                                            unsigned int tensor_d0,
                                            unsigned int tensor_d1,
                                            unsigned int tile_d0,
                                            unsigned int tile_d1,
                                            unsigned long long stride0) {
  unsigned long long ga = (unsigned long long)gptr;
  v4u g0;
  g0.x = 1u;                                           // count=1 (valid user D#)
  g0.y = (unsigned int)(unsigned long long)lptr;       // lds_addr
  g0.z = (unsigned int)ga;                             // global_addr[31:0]
  g0.w = (unsigned int)((ga >> 32) & 0x01ffffffu)      // global_addr[56:32]
         | (2u << 30);                                 // type=2 ("image")
  v8i g1;
  g1[0] = (int)((1u << 16)      // data_size = 1 -> 2-byte elements
              | (1u << 20)      // pad_enable
              | (3u << 22)      // pad_interval: 16 dwords
              | (3u << 25));    // pad_amount:   4 dwords
  g1[1] = (int)((tensor_d0 & 0xffffu) << 16);                  // tensor_dim0 lo
  g1[2] = (int)((tensor_d0 >> 16) | ((tensor_d1 & 0xffffu) << 16));
  g1[3] = (int)((tensor_d1 >> 16) | (tile_d0 << 16));          // tile_dim0
  g1[4] = (int)tile_d1;                                        // tile_dim1 (dim2=0)
  g1[5] = (int)(stride0 & 0xffffffffu);                        // dim0 stride lo
  g1[6] = (int)((stride0 >> 32) & 0xffffu);                    // stride hi (dim1 stride=0)
  g1[7] = 0;
  v4i z4 = {0, 0, 0, 0};
  v8i z8 = {0, 0, 0, 0, 0, 0, 0, 0};
  __builtin_amdgcn_tensor_load_to_lds(g0, g1, z4, z4, z8, 0);
}
#endif

// Fragment load: per CDNA5 16-bit 16x32 layout, lane (idx, h) needs K pairs
// kp = h*4..h*4+3 and 8+h*4..11+h*4.  Tile layout [idx][kp] (stride TSTR
// dwords, 80B rows) makes those two contiguous 16B runs -> 2x ds_load_b128.
__device__ __forceinline__ v16bf load_frag(const unsigned int* lds,
                                           int idx, int h) {
  const uint4* p = (const uint4*)__builtin_assume_aligned(
      &lds[idx * TSTR + h * 4], 16);
  uint4 lo = p[0];   // kp = h*4   .. h*4+3
  uint4 hi = p[2];   // kp = 8+h*4 .. 11+h*4
  v8u u = {lo.x, lo.y, lo.z, lo.w, hi.x, hi.y, hi.z, hi.w};
  return __builtin_bit_cast(v16bf, u);
}

// ---- pass 1: fused projection GEMM  Out[m,n] = bf16(X[m,:]·W[n,:] + b[n]) --

extern "C" __global__ void __launch_bounds__(256)
sha_proj_kernel(const float* __restrict__ X, const float* __restrict__ W,
                const float* __restrict__ bias, unsigned short* __restrict__ Out) {
  __shared__ __align__(16) unsigned int ldsA[2][MT * TSTR];  // 2 x 10240 B
  __shared__ __align__(16) unsigned int ldsB[2][NT * TSTR];  // 2 x 10240 B
  const int m0 = blockIdx.x * MT, n0 = blockIdx.y * NT;
  const int tid = threadIdx.x, wave = tid >> 5, lane = tid & 31;
  const int r = lane & 15, h = lane >> 4;
  const int sp = tid & 15;       // staging: k-pair (coalesced over 16 lanes)
  const int sr0 = tid >> 4;      // staging: base row, rows sr0 + 16t

  unsigned int regA[8], regB[8];
  auto loadRegs = [&](int k0) {
#pragma unroll
    for (int t = 0; t < 8; ++t) {
      int row = sr0 + t * 16;
      float2 fa = *(const float2*)(X + (size_t)(m0 + row) * DIM + k0 + sp * 2);
      regA[t] = pack2_bf16(fa.x, fa.y);
      float2 fb = *(const float2*)(W + (size_t)(n0 + row) * DIM + k0 + sp * 2);
      regB[t] = pack2_bf16(fb.x, fb.y);
    }
  };
  auto storeLds = [&](int buf) {
#pragma unroll
    for (int t = 0; t < 8; ++t) {
      int row = sr0 + t * 16;
      ldsA[buf][row * TSTR + sp] = regA[t];
      ldsB[buf][row * TSTR + sp] = regB[t];
    }
  };

  v8f acc[8] = {};
  loadRegs(0);
  storeLds(0);
  const int NKC = DIM / KT;
  for (int kc = 0; kc < NKC; ++kc) {
    __syncthreads();
    if (kc + 1 < NKC) loadRegs((kc + 1) * KT);
    const int buf = kc & 1;
    v16bf a = load_frag(ldsA[buf], wave * 16 + r, h);
#pragma unroll
    for (int c = 0; c < 8; ++c) {
      v16bf b = load_frag(ldsB[buf], c * 16 + r, h);
      acc[c] = __builtin_amdgcn_wmma_f32_16x16x32_bf16(
          false, a, false, b, (short)0, acc[c], false, false);
    }
    if (kc + 1 < NKC) storeLds(buf ^ 1);
  }

#pragma unroll
  for (int c = 0; c < 8; ++c) {
    int n = n0 + c * 16 + r;
    float bv = bias[n];
#pragma unroll
    for (int v = 0; v < 8; ++v) {
      int m = m0 + wave * 16 + v + 8 * h;
      Out[(size_t)m * DIM + n] = f2bf(acc[c][v] + bv);
    }
  }
}

// ---- pass 2: scores = scale*(Q·K^T), masked to -1e9, fp32 out --------------
// Staging: one TDM descriptor per 128x32 bf16 tile (wave0: Q, wave1: K),
// TENSORcnt-tracked; async-to-LDS fallback if the builtin is missing.

extern "C" __global__ void __launch_bounds__(256)
sha_scores_kernel(const unsigned short* __restrict__ Qw,
                  const unsigned short* __restrict__ Kw,
                  const unsigned char* __restrict__ mask,
                  float* __restrict__ Sc) {
  __shared__ __align__(16) unsigned int ldsA[2][MT * TSTR];
  __shared__ __align__(16) unsigned int ldsB[2][NT * TSTR];
  const int b = blockIdx.z;
  const unsigned short* Q16 = Qw + (size_t)b * SEQ * DIM;
  const unsigned short* K16 = Kw + (size_t)b * SEQ * DIM;
  const int m0 = blockIdx.x * MT, n0 = blockIdx.y * NT;
  const int tid = threadIdx.x, wave = tid >> 5, lane = tid & 31;
  const int r = lane & 15, h = lane >> 4;

#if HAS_TDM
  auto issueTiles = [&](int buf, int k0) {
    if (wave == 0)
      tdm_load_2d(&ldsA[buf][0], Q16 + (size_t)m0 * DIM + k0,
                  DIM, SEQ, KT, MT, DIM);
    else if (wave == 1)
      tdm_load_2d(&ldsB[buf][0], K16 + (size_t)n0 * DIM + k0,
                  DIM, SEQ, KT, NT, DIM);
  };
  auto waitTiles = [&]() {
    if (wave < 2) __builtin_amdgcn_s_wait_tensorcnt(0);
  };
#else
  const unsigned int* Q32 = (const unsigned int*)Q16;
  const unsigned int* K32 = (const unsigned int*)K16;
  const int qp = tid & 3;        // staging: 16B quad within row (coalesced)
  const int qr0 = tid >> 2;      // staging: base row, rows qr0 + 64t
  auto issueTiles = [&](int buf, int k0) {
#pragma unroll
    for (int t = 0; t < 2; ++t) {
      int qr = qr0 + t * 64;
      async_copy_b128(Q32 + (((size_t)(m0 + qr) * DIM + k0) >> 1) + qp * 4,
                      &ldsA[buf][qr * TSTR + qp * 4]);
      async_copy_b128(K32 + (((size_t)(n0 + qr) * DIM + k0) >> 1) + qp * 4,
                      &ldsB[buf][qr * TSTR + qp * 4]);
    }
  };
  auto waitTiles = [&]() { wait_async0(); };
#endif

  v8f acc[8] = {};
  issueTiles(0, 0);
  waitTiles();
  const int NKC = DIM / KT;
  for (int kc = 0; kc < NKC; ++kc) {
    __syncthreads();
    if (kc + 1 < NKC) issueTiles((kc + 1) & 1, (kc + 1) * KT);
    const int buf = kc & 1;
    v16bf a = load_frag(ldsA[buf], wave * 16 + r, h);
#pragma unroll
    for (int c = 0; c < 8; ++c) {
      v16bf b2 = load_frag(ldsB[buf], c * 16 + r, h);
      acc[c] = __builtin_amdgcn_wmma_f32_16x16x32_bf16(
          false, a, false, b2, (short)0, acc[c], false, false);
    }
    if (kc + 1 < NKC) waitTiles();   // LDS writes done before next barrier
  }

  const float scale = 0.03125f;  // 1/sqrt(1024)
#pragma unroll
  for (int c = 0; c < 8; ++c) {
    int n = n0 + c * 16 + r;
#pragma unroll
    for (int v = 0; v < 8; ++v) {
      int m = m0 + wave * 16 + v + 8 * h;
      size_t off = ((size_t)b * SEQ + m) * SEQ + n;
      float s = acc[c][v] * scale;
      if (mask[off]) s = -1.0e9f;   // jnp.where(mask, -NEG_INF, scores)
      Sc[off] = s;
    }
  }
}

// ---- pass 3: row softmax (fp32 in, bf16 P out) -----------------------------

extern "C" __global__ void __launch_bounds__(256)
sha_softmax_kernel(const float* __restrict__ Sc, unsigned short* __restrict__ P) {
  __shared__ float red[256];
  const size_t row = blockIdx.x;
  const float* srow = Sc + row * SEQ;
  const int tid = threadIdx.x;

  float x[8];
  float mx = -3.4e38f;
#pragma unroll
  for (int i = 0; i < 8; ++i) { x[i] = srow[tid + i * 256]; mx = fmaxf(mx, x[i]); }
  red[tid] = mx; __syncthreads();
  for (int s = 128; s > 0; s >>= 1) {
    if (tid < s) red[tid] = fmaxf(red[tid], red[tid + s]);
    __syncthreads();
  }
  mx = red[0]; __syncthreads();

  float sum = 0.0f;
#pragma unroll
  for (int i = 0; i < 8; ++i) { x[i] = __expf(x[i] - mx); sum += x[i]; }
  red[tid] = sum; __syncthreads();
  for (int s = 128; s > 0; s >>= 1) {
    if (tid < s) red[tid] += red[tid + s];
    __syncthreads();
  }
  float inv = 1.0f / red[0];
#pragma unroll
  for (int i = 0; i < 8; ++i)
    P[row * SEQ + tid + i * 256] = f2bf(x[i] * inv);
}

// ---- pass 4: Out = P·V (fp32 result) ---------------------------------------
// (A tile is a verbatim copy -> async-to-LDS; B needs k-pair interleave -> regs)

extern "C" __global__ void __launch_bounds__(256)
sha_out_kernel(const unsigned short* __restrict__ Pw,
               const unsigned short* __restrict__ Vw,
               float* __restrict__ Out) {
  __shared__ __align__(16) unsigned int ldsA[2][MT * TSTR];
  __shared__ __align__(16) unsigned int ldsB[2][NT * TSTR];
  const int b = blockIdx.z;
  const unsigned int* P32 = (const unsigned int*)(Pw + (size_t)b * SEQ * SEQ);
  const unsigned short* V = Vw + (size_t)b * SEQ * DIM;
  const int m0 = blockIdx.x * MT, n0 = blockIdx.y * NT;
  const int tid = threadIdx.x, wave = tid >> 5, lane = tid & 31;
  const int r = lane & 15, h = lane >> 4;
  const int qp = tid & 3;          // A staging: 16B quad
  const int qr0 = tid >> 2;        // A staging: base row, rows qr0 + 64t
  const int bn = (tid & 63) * 2;   // B staging: even column, fixed per thread
  const int bk0 = tid >> 6;        // B staging: key row base, rows bk0 + 4t

  auto issueAsyncA = [&](int buf, int k0) {
#pragma unroll
    for (int t = 0; t < 2; ++t) {
      int qr = qr0 + t * 64;
      async_copy_b128(P32 + (((size_t)(m0 + qr) * SEQ + k0) >> 1) + qp * 4,
                      &ldsA[buf][qr * TSTR + qp * 4]);
    }
  };
  unsigned int regB[8];
  auto loadRegsB = [&](int k0) {
#pragma unroll
    for (int t = 0; t < 8; ++t) {
      int kk = bk0 + t * 4;
      regB[t] = *(const unsigned int*)(V + (size_t)(k0 + kk) * DIM + n0 + bn);
    }
  };
  auto storeLdsB = [&](int buf) {
    // scatter V (row-major over N) into per-lane-contiguous [n][kpair] pairs
    unsigned short* lb = (unsigned short*)ldsB[buf];
#pragma unroll
    for (int t = 0; t < 8; ++t) {
      int kk = bk0 + t * 4;
      int kp = kk >> 1, sel = kk & 1;
      lb[(bn * TSTR + kp) * 2 + sel]       = (unsigned short)(regB[t] & 0xffffu);
      lb[((bn + 1) * TSTR + kp) * 2 + sel] = (unsigned short)(regB[t] >> 16);
    }
  };

  v8f acc[8] = {};
  issueAsyncA(0, 0);
  loadRegsB(0);
  storeLdsB(0);
  wait_async0();
  const int NKC = SEQ / KT;
  for (int kc = 0; kc < NKC; ++kc) {
    __syncthreads();
    if (kc + 1 < NKC) { issueAsyncA((kc + 1) & 1, (kc + 1) * KT); loadRegsB((kc + 1) * KT); }
    const int buf = kc & 1;
    v16bf a = load_frag(ldsA[buf], wave * 16 + r, h);
#pragma unroll
    for (int c = 0; c < 8; ++c) {
      v16bf b2 = load_frag(ldsB[buf], c * 16 + r, h);
      acc[c] = __builtin_amdgcn_wmma_f32_16x16x32_bf16(
          false, a, false, b2, (short)0, acc[c], false, false);
    }
    if (kc + 1 < NKC) { storeLdsB(buf ^ 1); wait_async0(); }
  }

#pragma unroll
  for (int c = 0; c < 8; ++c) {
    int n = n0 + c * 16 + r;
#pragma unroll
    for (int v = 0; v < 8; ++v) {
      int m = m0 + wave * 16 + v + 8 * h;
      Out[((size_t)b * SEQ + m) * DIM + n] = acc[c][v];
    }
  }
}

// ---- host-side launch ------------------------------------------------------

extern "C" void kernel_launch(void* const* d_in, const int* in_sizes, int n_in,
                              void* d_out, int out_size, void* d_ws, size_t ws_size,
                              hipStream_t stream) {
  (void)in_sizes; (void)n_in; (void)out_size; (void)ws_size;
  const float* query = (const float*)d_in[0];
  const float* key   = (const float*)d_in[1];
  const float* value = (const float*)d_in[2];
  const unsigned char* mask = (const unsigned char*)d_in[3];  // jax bool -> 1B
  const float* Wq = (const float*)d_in[4];
  const float* bq = (const float*)d_in[5];
  const float* Wk = (const float*)d_in[6];
  const float* bk = (const float*)d_in[7];
  const float* Wv = (const float*)d_in[8];
  const float* bv = (const float*)d_in[9];
  float* out = (float*)d_out;

  // workspace: Q,K,V bf16 (3x16.8MB) + scores f32 (67MB) + P bf16 (34MB)
  unsigned short* Qw = (unsigned short*)d_ws;
  unsigned short* Kw = Qw + (size_t)MROWS * DIM;
  unsigned short* Vw = Kw + (size_t)MROWS * DIM;
  float*          Sc = (float*)(Vw + (size_t)MROWS * DIM);
  unsigned short* Pw = (unsigned short*)(Sc + (size_t)BATCH * SEQ * SEQ);

  dim3 blk(256);
  dim3 gproj(MROWS / MT, DIM / NT, 1);
  sha_proj_kernel<<<gproj, blk, 0, stream>>>(query, Wq, bq, Qw);
  sha_proj_kernel<<<gproj, blk, 0, stream>>>(key,   Wk, bk, Kw);
  sha_proj_kernel<<<gproj, blk, 0, stream>>>(value, Wv, bv, Vw);

  dim3 gsc(SEQ / MT, SEQ / NT, BATCH);
  sha_scores_kernel<<<gsc, blk, 0, stream>>>(Qw, Kw, mask, Sc);

  sha_softmax_kernel<<<dim3(MROWS), blk, 0, stream>>>(Sc, Pw);

  dim3 gout(SEQ / MT, DIM / NT, BATCH);
  sha_out_kernel<<<gout, blk, 0, stream>>>(Pw, Vw, out);
}
